// RQVAE_58849641890162
// MI455X (gfx1250) — matrix operand order repaired and checked by reading, not compiled, not run on hardware
//
#include <hip/hip_runtime.h>
#include <cstdint>

// ---------------------------------------------------------------------------
// RQ-VAE forward for MI455X (gfx1250, wave32, WMMA f32<-f16 16x16x32)
//  - all 6 MLP GEMMs via v_wmma_f32_16x16x32_f16, weights pre-transposed to
//    N-major f16 so B-frags load straight from L2 (no LDS transpose)
//  - A tiles double-buffered in LDS; async global->LDS copies via inline asm
//    (GLOBAL_LOAD_ASYNC_TO_LDS_B128 + s_wait_asynccnt, ISA 15.18.3 / ch.5)
//  - residual quantizer: WMMA distance GEMM + shuffle argmax per level
// ---------------------------------------------------------------------------

typedef _Float16 v8h  __attribute__((ext_vector_type(8)));
typedef _Float16 v16h __attribute__((ext_vector_type(16)));
typedef _Float16 h4   __attribute__((ext_vector_type(4)));
typedef float    v8f  __attribute__((ext_vector_type(8)));

#define USE_ASYNC_LDS 1

__device__ __forceinline__ void async_copy16(const void* g, void* lds) {
  // LDS byte offset = low 32 bits of the flat LDS address (aperture rules).
  unsigned l32 = (unsigned)(uintptr_t)lds;
  unsigned long long g64 = (unsigned long long)(uintptr_t)g;
  asm volatile("global_load_async_to_lds_b128 %0, %1, off"
               :
               : "v"(l32), "v"(g64)
               : "memory");
}

__device__ __forceinline__ void wait_async0() {
#if defined(__has_builtin) && __has_builtin(__builtin_amdgcn_s_wait_asynccnt)
  __builtin_amdgcn_s_wait_asynccnt(0);
#else
  asm volatile("s_wait_asynccnt 0x0" ::: "memory");
#endif
}

__device__ __forceinline__ v16h cat8(v8h a, v8h b) {
  return __builtin_shufflevector(a, b, 0,1,2,3,4,5,6,7,8,9,10,11,12,13,14,15);
}

__device__ __forceinline__ v8f wmma16(v16h a, v16h b, v8f c) {
  // D = A(16x32 f16) * B(32x16 f16) + C(16x16 f32)
  return __builtin_amdgcn_wmma_f32_16x16x32_f16(false, a, false, b, (short)0, c,
                                                false, false);
}

// ---------------------------------------------------------------------------
// conversions
// ---------------------------------------------------------------------------
__global__ void f32_to_f16_kernel(const float* __restrict__ src,
                                  _Float16* __restrict__ dst, int n) {
  int i = blockIdx.x * 256 + threadIdx.x;
  if (i < n) dst[i] = (_Float16)src[i];
}

// W[K,N] f32 -> Wt[N,K] f16
__global__ void f32_to_f16_t_kernel(const float* __restrict__ src,
                                    _Float16* __restrict__ dst, int K, int N) {
  int i = blockIdx.x * 256 + threadIdx.x;
  if (i < K * N) {
    int k = i / N, n = i - k * N;
    dst[(size_t)n * K + k] = (_Float16)src[i];
  }
}

// c2[j] = sum_d C[j][d]^2 for all 4*256 codes; also zero the loss accumulator.
__global__ void c2_init_kernel(const float* __restrict__ cb,
                               float* __restrict__ c2,
                               float* __restrict__ loss_acc) {
  int j = blockIdx.x * 256 + threadIdx.x;  // 0..1023
  if (j < 1024) {
    const float* p = cb + (size_t)j * 128;
    float s = 0.f;
    for (int d = 0; d < 128; ++d) { float v = p[d]; s += v * v; }
    c2[j] = s;
  }
  if (j == 0) *loss_acc = 0.f;
}

// ---------------------------------------------------------------------------
// GEMM: C[M,N] = act(A[M,K] @ W[K,N] + bias[N]),  W given as Wt[N,K] f16.
//   block tile 128(M) x 128(N); 8 waves, each 64x32 (4x2 WMMA frags).
//   A tile (128x64 f16) double-buffered in LDS (async copy for f16 A);
//   B frags read directly from global Wt (L2-resident weights).
// ---------------------------------------------------------------------------
template <typename TA, typename TO, bool RELU>
__global__ __launch_bounds__(256)
void mlp_gemm_kernel(const TA* __restrict__ A, const _Float16* __restrict__ Wt,
                     const float* __restrict__ bias, TO* __restrict__ C,
                     int K, int N) {
  __shared__ _Float16 As[2][128][72];  // row stride 144B = 9*16B

  const int tid  = threadIdx.x;
  const int lane = tid & 31;
  const int wv   = tid >> 5;      // wave 0..7
  const int wm   = wv >> 2;       // 0..1 -> M offset wm*64
  const int wn   = wv & 3;        // 0..3 -> N offset wn*32
  const int sel  = lane >> 4;
  const int lo   = lane & 15;

  const int bm = blockIdx.y * 128;
  const int bn = blockIdx.x * 128;
  const int nk = K >> 6;

  constexpr bool A_IS_F16 = (sizeof(TA) == 2);

  v8f acc[4][2] = {};
  h4  stage[8];

  auto load_regs = [&](int kt) {
#pragma unroll
    for (int i = 0; i < 8; ++i) {
      int idx = i * 256 + tid;  // 2048 h4 slots of the 128x64 tile
      int row = idx >> 4;
      int c4  = (idx & 15) << 2;
      if constexpr (A_IS_F16) {
        stage[i] = *(const h4*)((const _Float16*)A +
                                (size_t)(bm + row) * K + kt * 64 + c4);
      } else {
        const float4 v = *(const float4*)((const float*)A +
                                          (size_t)(bm + row) * K + kt * 64 + c4);
        h4 hv;
        hv[0] = (_Float16)v.x; hv[1] = (_Float16)v.y;
        hv[2] = (_Float16)v.z; hv[3] = (_Float16)v.w;
        stage[i] = hv;
      }
    }
  };
  auto store_lds = [&](int b) {
#pragma unroll
    for (int i = 0; i < 8; ++i) {
      int idx = i * 256 + tid;
      int row = idx >> 4;
      int c4  = (idx & 15) << 2;
      *(h4*)&As[b][row][c4] = stage[i];
    }
  };

#if USE_ASYNC_LDS
  auto stage_async = [&](int kt, int b) {
#pragma unroll
    for (int i = 0; i < 4; ++i) {
      int c   = i * 256 + tid;  // 1024 16-byte chunks of the 128x64 tile
      int row = c >> 3;
      int h8  = (c & 7) << 3;
      const _Float16* g =
          (const _Float16*)A + (size_t)(bm + row) * K + kt * 64 + h8;
      async_copy16((const void*)g, (void*)&As[b][row][h8]);
    }
  };
#endif

  auto compute = [&](int b, int kt) {
#pragma unroll
    for (int ks = 0; ks < 2; ++ks) {
      const int kb = ks * 32;
      v16h a[4], bf[2];
#pragma unroll
      for (int mf = 0; mf < 4; ++mf) {
        const int mrow = wm * 64 + mf * 16 + lo;
        v8h a0 = *(const v8h*)&As[b][mrow][kb + 16 * sel];
        v8h a1 = *(const v8h*)&As[b][mrow][kb + 16 * sel + 8];
        a[mf]  = cat8(a0, a1);
      }
#pragma unroll
      for (int nf = 0; nf < 2; ++nf) {
        const _Float16* cp = Wt + (size_t)(bn + wn * 32 + nf * 16 + lo) * K +
                             kt * 64 + kb + 16 * sel;
        v8h b0 = *(const v8h*)cp;
        v8h b1 = *(const v8h*)(cp + 8);
        bf[nf] = cat8(b0, b1);
      }
#pragma unroll
      for (int mf = 0; mf < 4; ++mf)
#pragma unroll
        for (int nf = 0; nf < 2; ++nf)
          acc[mf][nf] = wmma16(a[mf], bf[nf], acc[mf][nf]);
    }
  };

#if USE_ASYNC_LDS
  if constexpr (A_IS_F16) {
    stage_async(0, 0);
    wait_async0();
    __syncthreads();
    for (int kt = 0; kt < nk; ++kt) {
      const int b = kt & 1;
      if (kt + 1 < nk) stage_async(kt + 1, b ^ 1);
      compute(b, kt);
      if (kt + 1 < nk) wait_async0();
      __syncthreads();
    }
  } else
#endif
  {
    load_regs(0);
    store_lds(0);
    __syncthreads();
    for (int kt = 0; kt < nk; ++kt) {
      const int b = kt & 1;
      if (kt + 1 < nk) load_regs(kt + 1);
      compute(b, kt);
      if (kt + 1 < nk) store_lds(b ^ 1);
      __syncthreads();
    }
  }

  // ---- epilogue: bias (+ReLU) and store ----
#pragma unroll
  for (int mf = 0; mf < 4; ++mf) {
#pragma unroll
    for (int nf = 0; nf < 2; ++nf) {
      const int col = bn + wn * 32 + nf * 16 + lo;
      const float bv = bias[col];
#pragma unroll
      for (int r = 0; r < 8; ++r) {
        const int rowg = bm + wm * 64 + mf * 16 + sel * 8 + r;
        float v = acc[mf][nf][r] + bv;
        if (RELU) v = fmaxf(v, 0.f);
        C[(size_t)rowg * N + col] = (TO)v;
      }
    }
  }
}

// ---------------------------------------------------------------------------
// Residual quantizer: 4 levels over z[B,128] with codebooks[4,256,128].
// ---------------------------------------------------------------------------
__global__ __launch_bounds__(256)
void rq_kernel(const float* __restrict__ z, const _Float16* __restrict__ cbh,
               const float* __restrict__ cb, const float* __restrict__ c2,
               _Float16* __restrict__ yq, float* __restrict__ loss_acc) {
  __shared__ float    zr[64][129];
  __shared__ _Float16 rh[64][136];  // 272B row stride, 16B-aligned
  __shared__ float    candv[8][64];
  __shared__ int      candi[8][64];
  __shared__ int      idxs[64];
  __shared__ float    red[256];

  const int tid  = threadIdx.x;
  const int lane = tid & 31;
  const int wv   = tid >> 5;
  const int sel  = lane >> 4;
  const int lo   = lane & 15;
  const size_t base = (size_t)blockIdx.x * 64 * 128;

  for (int it = 0; it < 8; ++it) {
    int idx = it * 256 + tid;
    int row = idx >> 5;
    int c4  = (idx & 31) << 2;
    const float4 v = *(const float4*)(z + base + (size_t)row * 128 + c4);
    zr[row][c4 + 0] = v.x; zr[row][c4 + 1] = v.y;
    zr[row][c4 + 2] = v.z; zr[row][c4 + 3] = v.w;
    rh[row][c4 + 0] = (_Float16)v.x; rh[row][c4 + 1] = (_Float16)v.y;
    rh[row][c4 + 2] = (_Float16)v.z; rh[row][c4 + 3] = (_Float16)v.w;
  }

  float lsum = 0.f;

  for (int l = 0; l < 4; ++l) {
    __syncthreads();

    v8f s[4][2] = {};
    const _Float16* cbl = cbh + (size_t)l * 256 * 128;
#pragma unroll
    for (int ks = 0; ks < 4; ++ks) {
      const int kb = ks * 32;
      v16h a[4], b[2];
#pragma unroll
      for (int mf = 0; mf < 4; ++mf) {
        const int mrow = mf * 16 + lo;
        v8h a0 = *(const v8h*)&rh[mrow][kb + 16 * sel];
        v8h a1 = *(const v8h*)&rh[mrow][kb + 16 * sel + 8];
        a[mf]  = cat8(a0, a1);
      }
#pragma unroll
      for (int nf = 0; nf < 2; ++nf) {
        const int code = wv * 32 + nf * 16 + lo;
        const _Float16* cp = cbl + (size_t)code * 128 + kb + 16 * sel;
        v8h b0 = *(const v8h*)cp;
        v8h b1 = *(const v8h*)(cp + 8);
        b[nf]  = cat8(b0, b1);
      }
#pragma unroll
      for (int mf = 0; mf < 4; ++mf)
#pragma unroll
        for (int nf = 0; nf < 2; ++nf)
          s[mf][nf] = wmma16(a[mf], b[nf], s[mf][nf]);
    }

    const int code0 = wv * 32 + lo;
    const int code1 = wv * 32 + 16 + lo;
    const float h0  = 0.5f * c2[l * 256 + code0];
    const float h1c = 0.5f * c2[l * 256 + code1];
#pragma unroll
    for (int mf = 0; mf < 4; ++mf) {
#pragma unroll
      for (int r = 0; r < 8; ++r) {
        float v0 = s[mf][0][r] - h0;
        float v1 = s[mf][1][r] - h1c;
        float bvv; int bii;
        if (v1 > v0) { bvv = v1; bii = code1; } else { bvv = v0; bii = code0; }
#pragma unroll
        for (int m = 1; m <= 8; m <<= 1) {
          float ov = __shfl_xor(bvv, m, 32);
          int   oi = __shfl_xor(bii, m, 32);
          if (ov > bvv) { bvv = ov; bii = oi; }
        }
        if (lo == 0) {
          const int row = mf * 16 + sel * 8 + r;
          candv[wv][row] = bvv;
          candi[wv][row] = bii;
        }
      }
    }
    __syncthreads();

    if (tid < 64) {
      float bvv = candv[0][tid];
      int   bii = candi[0][tid];
      for (int ww = 1; ww < 8; ++ww) {
        if (candv[ww][tid] > bvv) { bvv = candv[ww][tid]; bii = candi[ww][tid]; }
      }
      idxs[tid] = bii;
    }
    __syncthreads();

    const float* cl = cb + (size_t)l * 256 * 128;
    for (int it = 0; it < 8; ++it) {
      int idx = it * 256 + tid;
      int row = idx >> 5;
      int c4  = (idx & 31) << 2;
      const float4 q = *(const float4*)(cl + (size_t)idxs[row] * 128 + c4);
#pragma unroll
      for (int j = 0; j < 4; ++j) {
        const float qv = (j == 0) ? q.x : (j == 1) ? q.y : (j == 2) ? q.z : q.w;
        float ro = zr[row][c4 + j];
        float df = qv - ro;
        lsum += df * df;
        ro -= qv;
        zr[row][c4 + j] = ro;
        rh[row][c4 + j] = (_Float16)ro;
      }
    }
  }
  __syncthreads();

  for (int it = 0; it < 8; ++it) {
    int idx = it * 256 + tid;
    int row = idx >> 5;
    int c4  = (idx & 31) << 2;
    const float4 zv = *(const float4*)(z + base + (size_t)row * 128 + c4);
    h4 o;
    o[0] = (_Float16)(zv.x - zr[row][c4 + 0]);
    o[1] = (_Float16)(zv.y - zr[row][c4 + 1]);
    o[2] = (_Float16)(zv.z - zr[row][c4 + 2]);
    o[3] = (_Float16)(zv.w - zr[row][c4 + 3]);
    *(h4*)(yq + base + (size_t)row * 128 + c4) = o;
  }

  red[tid] = lsum;
  __syncthreads();
  for (int s2 = 128; s2 > 0; s2 >>= 1) {
    if (tid < s2) red[tid] += red[tid + s2];
    __syncthreads();
  }
  if (tid == 0) atomicAdd(loss_acc, red[0]);
}

__global__ void finalize_kernel(const float* __restrict__ loss_acc,
                                float* __restrict__ out) {
  // rq_loss = (1/4) * sum_l (1+beta) * sumsq_l / (B*128), beta = 0.25
  *out = (*loss_acc) * (1.25f / (4.0f * 65536.0f * 128.0f));
}

// ---------------------------------------------------------------------------
// Host side
// ---------------------------------------------------------------------------
extern "C" void kernel_launch(void* const* d_in, const int* in_sizes, int n_in,
                              void* d_out, int out_size, void* d_ws,
                              size_t ws_size, hipStream_t stream) {
  const float* x   = (const float*)d_in[0];
  const float* ew0 = (const float*)d_in[1];
  const float* eb0 = (const float*)d_in[2];
  const float* ew1 = (const float*)d_in[3];
  const float* eb1 = (const float*)d_in[4];
  const float* ew2 = (const float*)d_in[5];
  const float* eb2 = (const float*)d_in[6];
  const float* dw0 = (const float*)d_in[7];
  const float* db0 = (const float*)d_in[8];
  const float* dw1 = (const float*)d_in[9];
  const float* db1 = (const float*)d_in[10];
  const float* dw2 = (const float*)d_in[11];
  const float* db2 = (const float*)d_in[12];
  const float* cb  = (const float*)d_in[13];

  const int B = 65536;
  char* wsb = (char*)d_ws;
  size_t off = 0;
  auto alloc = [&](size_t bytes) -> void* {
    void* p = wsb + off;
    off = (off + bytes + 255) & ~(size_t)255;
    return p;
  };

  float*    loss = (float*)alloc(4);
  float*    c2   = (float*)alloc(1024 * 4);
  _Float16* W0t  = (_Float16*)alloc((size_t)768 * 512 * 2);
  _Float16* W1t  = (_Float16*)alloc((size_t)512 * 256 * 2);
  _Float16* W2t  = (_Float16*)alloc((size_t)256 * 128 * 2);
  _Float16* D0t  = (_Float16*)alloc((size_t)128 * 256 * 2);
  _Float16* D1t  = (_Float16*)alloc((size_t)256 * 512 * 2);
  _Float16* D2t  = (_Float16*)alloc((size_t)512 * 768 * 2);
  _Float16* cbh  = (_Float16*)alloc((size_t)4 * 256 * 128 * 2);
  _Float16* h1   = (_Float16*)alloc((size_t)B * 512 * 2);
  _Float16* h2   = (_Float16*)alloc((size_t)B * 256 * 2);
  float*    zb   = (float*)alloc((size_t)B * 128 * 4);
  _Float16* yq   = (_Float16*)alloc((size_t)B * 128 * 2);
  _Float16* d1   = h2;  // reuse (h2 dead after encoder GEMM 3)
  _Float16* d2   = h1;  // reuse (h1 dead after encoder GEMM 2)

  auto cvtT = [&](const float* s, _Float16* d, int K, int N) {
    f32_to_f16_t_kernel<<<dim3((K * N + 255) / 256), dim3(256), 0, stream>>>(
        s, d, K, N);
  };
  cvtT(ew0, W0t, 768, 512);
  cvtT(ew1, W1t, 512, 256);
  cvtT(ew2, W2t, 256, 128);
  cvtT(dw0, D0t, 128, 256);
  cvtT(dw1, D1t, 256, 512);
  cvtT(dw2, D2t, 512, 768);
  f32_to_f16_kernel<<<dim3((4 * 256 * 128 + 255) / 256), dim3(256), 0, stream>>>(
      cb, cbh, 4 * 256 * 128);
  c2_init_kernel<<<dim3(4), dim3(256), 0, stream>>>(cb, c2, loss);

  const dim3 blk(256);
  // encoder
  mlp_gemm_kernel<float, _Float16, true>
      <<<dim3(512 / 128, B / 128), blk, 0, stream>>>(x, W0t, eb0, h1, 768, 512);
  mlp_gemm_kernel<_Float16, _Float16, true>
      <<<dim3(256 / 128, B / 128), blk, 0, stream>>>(h1, W1t, eb1, h2, 512, 256);
  mlp_gemm_kernel<_Float16, float, false>
      <<<dim3(128 / 128, B / 128), blk, 0, stream>>>(h2, W2t, eb2, zb, 256, 128);
  // residual quantization
  rq_kernel<<<dim3(B / 64), blk, 0, stream>>>(zb, cbh, cb, c2, yq, loss);
  // decoder
  mlp_gemm_kernel<_Float16, _Float16, true>
      <<<dim3(256 / 128, B / 128), blk, 0, stream>>>(yq, D0t, db0, d1, 128, 256);
  mlp_gemm_kernel<_Float16, _Float16, true>
      <<<dim3(512 / 128, B / 128), blk, 0, stream>>>(d1, D1t, db1, d2, 256, 512);
  mlp_gemm_kernel<_Float16, float, false>
      <<<dim3(768 / 128, B / 128), blk, 0, stream>>>(d2, D2t, db2,
                                                     (float*)d_out, 512, 768);
  // scalar loss -> last output element
  finalize_kernel<<<dim3(1), dim3(1), 0, stream>>>(
      loss, (float*)d_out + (size_t)B * 768);
}
